// TransformerBlock_8186207666352
// MI455X (gfx1250) — compile-verified, matrix operand order checked by reading, not run on hardware
//
#include <hip/hip_runtime.h>
#include <hip/hip_bf16.h>

#define DEVINL __device__ __forceinline__

typedef __bf16 bf16;
typedef __attribute__((ext_vector_type(16))) __bf16 v16bf;
typedef __attribute__((ext_vector_type(8)))  float  v8f;
typedef __attribute__((ext_vector_type(4)))  unsigned int u32x4;
typedef __attribute__((ext_vector_type(8)))  int  i32x8;
typedef __attribute__((ext_vector_type(4)))  int  i32x4;

static constexpr int NN = 32768;   // rows
static constexpr int GG = 128;     // ensemble group size
static constexpr int EE = 512;     // embed dim
static constexpr int HH = 8;       // heads (d=64)

#define GEMM_USE_TDM 1
#if __has_include(<hip/amd_detail/amd_gfx1250_TDM.h>)
#define TDM_6ARG 1                 // therock-10.0 headers -> 6-arg builtin
#else
#define TDM_6ARG 0                 // ROCm 7.2 -> 5-arg builtin
#endif

union BF8 { uint4 u4; bf16 b[8]; };

DEVINL v8f v8f_zero() {
  v8f z = {0.f,0.f,0.f,0.f,0.f,0.f,0.f,0.f};
  return z;
}

// ---------------------------------------------------------------------------
// WMMA fragment helpers (wave32, 16x16x32 bf16).
// A-matrix 16x32 bf16: lanes 0-15 row M=lane hold K=0..7 (v0..3, packed pairs)
// and K=16..23 (v4..7); lanes 16-31 same rows, K=8..15 / K=24..31. The B
// operand is W[n][k] row-major which matches the same per-lane pattern.
// ---------------------------------------------------------------------------
DEVINL v16bf load_frag(const bf16* __restrict__ p, int ld) {
  const int lane = threadIdx.x & 31;
  const int r  = lane & 15;
  const int kb = (lane >> 4) << 3;          // 0 or 8
  const bf16* row = p + r * ld + kb;
  union { v16bf v; unsigned u[8]; } f;
  const unsigned* lo = (const unsigned*)(row);
  const unsigned* hi = (const unsigned*)(row + 16);
#pragma unroll
  for (int i = 0; i < 4; ++i) { f.u[i] = lo[i]; f.u[4 + i] = hi[i]; }
  return f.v;
}

DEVINL v8f wmma_bf16(v16bf a, v16bf b, v8f c) {
  return __builtin_amdgcn_wmma_f32_16x16x32_bf16(false, a, false, b, (short)0, c,
                                                 false, false);
}

// ---------------------------------------------------------------------------
// Tensor Data Mover: 2D strided bf16 tile (global -> LDS), D# per ISA ch.8.
// Tensor dims set huge (tiles always fully in-bounds). Only the global
// address words vary per issue; the rest is loop-invariant and hoistable.
// ---------------------------------------------------------------------------
DEVINL void tdm_issue_2d(unsigned lds_off, const bf16* gptr,
                         unsigned tile_h, unsigned stride) {
  const unsigned long long ga = (unsigned long long)gptr;
  u32x4 g0;
  g0.x = 1u;                                     // count=1, user descriptor
  g0.y = lds_off;                                // LDS byte address
  g0.z = (unsigned)ga;                           // global_addr[31:0]
  g0.w = (unsigned)((ga >> 32) & 0x01FFFFFFu)    // global_addr[56:32]
         | (2u << 30);                           // type = 2 ("image")
  const unsigned td0 = 1u << 30, td1 = 1u << 30;
  i32x8 g1;
  g1[0] = (int)(1u << 16);                       // data_size = 1 -> 2 bytes
  g1[1] = (int)((td0 & 0xFFFFu) << 16);          // tensor_dim0[15:0]
  g1[2] = (int)((td0 >> 16) | ((td1 & 0xFFFFu) << 16));
  g1[3] = (int)((td1 >> 16) | (32u << 16));      // tile_dim0 = 32 elems
  g1[4] = (int)(tile_h & 0xFFFFu);               // tile_dim1 (tile_dim2=0)
  g1[5] = (int)stride;                           // tensor_dim0_stride[31:0]
  g1[6] = 0;
  g1[7] = 0;
  i32x4 z4 = {0, 0, 0, 0};
#if TDM_6ARG
  i32x8 z8 = {0, 0, 0, 0, 0, 0, 0, 0};
  __builtin_amdgcn_tensor_load_to_lds(g0, g1, z4, z4, z8, 0);
#else
  __builtin_amdgcn_tensor_load_to_lds(g0, g1, z4, z4, 0);
#endif
}

// ---------------------------------------------------------------------------
// f32 -> bf16 weight conversion (4 elems/thread)
// ---------------------------------------------------------------------------
__global__ __launch_bounds__(256) void cvt_kernel(const float* __restrict__ in,
                                                  bf16* __restrict__ out, int n4) {
  int i = blockIdx.x * 256 + threadIdx.x;
  if (i < n4) {
    float4 f = ((const float4*)in)[i];
    union { bf16 b[4]; unsigned long long u; } t;
    t.b[0] = (bf16)f.x; t.b[1] = (bf16)f.y; t.b[2] = (bf16)f.z; t.b[3] = (bf16)f.w;
    *(unsigned long long*)&out[(size_t)i * 4] = t.u;
  }
}

// ---------------------------------------------------------------------------
// features[:, :512] -> contiguous f32 x (one float4 per thread)
// ---------------------------------------------------------------------------
__global__ __launch_bounds__(256) void slice_kernel(const float* __restrict__ in,
                                                    float* __restrict__ out) {
  size_t g   = (size_t)blockIdx.x * 256 + threadIdx.x;
  size_t row = g >> 7;            // 128 float4 per row
  size_t c4  = g & 127;
  ((float4*)out)[row * 128 + c4] = ((const float4*)in)[row * 224 + c4]; // 896/4
}

// ---------------------------------------------------------------------------
// LayerNorm over E=512, one 256-thread block per row, bf16 output
// ---------------------------------------------------------------------------
__global__ __launch_bounds__(256) void ln_kernel(const float* __restrict__ x,
                                                 const float* __restrict__ g,
                                                 const float* __restrict__ bta,
                                                 bf16* __restrict__ out) {
  const int m = blockIdx.x;
  const int c = threadIdx.x * 2;
  float2 v = *(const float2*)&x[(size_t)m * EE + c];
  float s  = v.x + v.y;
  float sq = v.x * v.x + v.y * v.y;
#pragma unroll
  for (int o = 1; o < 32; o <<= 1) { s += __shfl_xor(s, o); sq += __shfl_xor(sq, o); }
  __shared__ float ss[8], qs[8];
  const int w = threadIdx.x >> 5;
  if ((threadIdx.x & 31) == 0) { ss[w] = s; qs[w] = sq; }
  __syncthreads();
  float ts = 0.f, tq = 0.f;
#pragma unroll
  for (int i = 0; i < 8; ++i) { ts += ss[i]; tq += qs[i]; }
  const float mean = ts * (1.f / EE);
  const float var  = tq * (1.f / EE) - mean * mean;
  const float inv  = rsqrtf(var + 1e-5f);
  union { bf16 b[2]; unsigned u; } t;
  t.b[0] = (bf16)((v.x - mean) * inv * g[c]     + bta[c]);
  t.b[1] = (bf16)((v.y - mean) * inv * g[c + 1] + bta[c + 1]);
  *(unsigned*)&out[(size_t)m * EE + c] = t.u;
}

// ---------------------------------------------------------------------------
// GEMM: C[M,N] = act(A[M,K] @ W[N,K]^T + bias) + resScale*resid
// Block tile 128x256, BK=32, 8 waves (2m x 4n), each wave 64x64 (16 v_wmma
// per K-step). Tiles staged by the Tensor Data Mover with double-buffered
// LDS: wave 0 issues tile t+1's two descriptors, then s_wait_tensorcnt(2)
// guarantees tile t is resident (TDM ops of a wave complete in order) while
// tile t+1's DMA overlaps the 16 WMMAs of tile t.
// ---------------------------------------------------------------------------
template <int ACT>   // 0 = none, 1 = SiLU
__global__ __launch_bounds__(256) void gemm_bf16_kernel(
    const bf16* __restrict__ A, int lda,
    const bf16* __restrict__ W, int ldw,
    const float* __restrict__ bias,
    const float* __restrict__ resid, float resScale,
    float* __restrict__ outF, bf16* __restrict__ outB, int ldo, int K) {
  __shared__ __align__(16) bf16 As[2][128 * 32];   // 2 x  8 KB
  __shared__ __align__(16) bf16 Bs[2][256 * 32];   // 2 x 16 KB

  const int tid  = threadIdx.x;
  const int w    = tid >> 5;
  const int wm   = w >> 2;          // 0..1 -> 64-row strip
  const int wn   = w & 3;           // 0..3 -> 64-col strip
  const int row0 = blockIdx.y * 128;
  const int col0 = blockIdx.x * 256;

  const bf16* pA = A + (size_t)row0 * lda;
  const bf16* pW = W + (size_t)col0 * ldw;

  v8f acc[4][4];
#pragma unroll
  for (int i = 0; i < 4; ++i)
#pragma unroll
    for (int j = 0; j < 4; ++j) acc[i][j] = v8f_zero();

  const int T = K >> 5;             // number of 32-wide K tiles

#if GEMM_USE_TDM
  const unsigned ldsA0 = (unsigned)(size_t)&As[0][0];
  const unsigned ldsA1 = (unsigned)(size_t)&As[1][0];
  const unsigned ldsB0 = (unsigned)(size_t)&Bs[0][0];
  const unsigned ldsB1 = (unsigned)(size_t)&Bs[1][0];
  if (w == 0) {                     // prologue: tile 0 -> buffer 0
    tdm_issue_2d(ldsA0, pA, 128u, (unsigned)lda);
    tdm_issue_2d(ldsB0, pW, 256u, (unsigned)ldw);
  }
#endif

  for (int t = 0; t < T; ++t) {
    const int cur = t & 1;
#if GEMM_USE_TDM
    __syncthreads();                // readers of buffer cur^1 (tile t-1) done
    if (w == 0) {
      if (t + 1 < T) {              // prefetch tile t+1 into the free buffer
        tdm_issue_2d(cur ? ldsA0 : ldsA1, pA + (t + 1) * 32, 128u, (unsigned)lda);
        tdm_issue_2d(cur ? ldsB0 : ldsB1, pW + (t + 1) * 32, 256u, (unsigned)ldw);
        __builtin_amdgcn_s_wait_tensorcnt(2);   // tile t resident, t+1 in flight
      } else {
        __builtin_amdgcn_s_wait_tensorcnt(0);
      }
    }
    __syncthreads();                // release compute on buffer cur
#else
    __syncthreads();
    for (int c = tid; c < 512; c += 256) {
      const int r = c >> 2, col = (c & 3) << 3;
      *(uint4*)&As[cur][r * 32 + col] =
          *(const uint4*)&pA[(size_t)r * lda + t * 32 + col];
    }
    for (int c = tid; c < 1024; c += 256) {
      const int r = c >> 2, col = (c & 3) << 3;
      *(uint4*)&Bs[cur][r * 32 + col] =
          *(const uint4*)&pW[(size_t)r * ldw + t * 32 + col];
    }
    __syncthreads();
#endif

    v16bf a[4], b[4];
#pragma unroll
    for (int i = 0; i < 4; ++i)
      a[i] = load_frag(&As[cur][(wm * 64 + i * 16) * 32], 32);
#pragma unroll
    for (int j = 0; j < 4; ++j)
      b[j] = load_frag(&Bs[cur][(wn * 64 + j * 16) * 32], 32);
#pragma unroll
    for (int i = 0; i < 4; ++i)
#pragma unroll
      for (int j = 0; j < 4; ++j) acc[i][j] = wmma_bf16(a[i], b[j], acc[i][j]);
  }

  // epilogue: C frag -> lane n (0-15) holds rows mb..mb+7 at col n
  const int lane = tid & 31;
  const int nl   = lane & 15;
  const int mb   = (lane >> 4) << 3;
#pragma unroll
  for (int i = 0; i < 4; ++i) {
#pragma unroll
    for (int j = 0; j < 4; ++j) {
      const int gn   = col0 + wn * 64 + j * 16 + nl;
      const float bv = bias ? bias[gn] : 0.f;
#pragma unroll
      for (int r = 0; r < 8; ++r) {
        const int gm = row0 + wm * 64 + i * 16 + mb + r;
        float v = acc[i][j][r] + bv;
        if (ACT == 1) v = v * (1.f / (1.f + __expf(-v)));   // SiLU
        const size_t o = (size_t)gm * ldo + gn;
        if (resid) v = fmaf(resScale, resid[o], v);
        if (outF) outF[o] = v;
        if (outB) outB[o] = (bf16)v;
      }
    }
  }
}

// ---------------------------------------------------------------------------
// Attention: one block per (ensemble, head). Full 16x128 score strip per
// wave in registers, softmax via half-wave shfl_xor reductions, P + V^T in
// LDS for the P@V WMMA pass.
// ---------------------------------------------------------------------------
__global__ __launch_bounds__(256) void attn_kernel(const bf16* __restrict__ Qg,
                                                   const bf16* __restrict__ Kg,
                                                   const bf16* __restrict__ Vg,
                                                   bf16* __restrict__ Og) {
  __shared__ __align__(16) bf16 Ks[128 * 64];   // [key][d]
  __shared__ __align__(16) bf16 Vt[64 * 128];   // [d][key]
  __shared__ __align__(16) bf16 Ps[128 * 128];  // probs

  const int    hb   = blockIdx.x;
  const size_t base = (size_t)(hb >> 3) * GG;
  const int    co   = (hb & 7) * 64;
  const int    tid  = threadIdx.x;

  for (int c = tid; c < 1024; c += 256) {       // 128x64 bf16 = 1024 uint4
    const int r   = c >> 3;
    const int col = (c & 7) << 3;
    *(uint4*)&Ks[r * 64 + col] = *(const uint4*)&Kg[(base + r) * EE + co + col];
    BF8 t; t.u4 = *(const uint4*)&Vg[(base + r) * EE + co + col];
#pragma unroll
    for (int i = 0; i < 8; ++i) Vt[(col + i) * 128 + r] = t.b[i];
  }
  __syncthreads();

  const int w    = tid >> 5;
  const int lane = tid & 31;

  // S = Q @ K^T for rows [w*16, w*16+16)
  v8f s[8];
#pragma unroll
  for (int j = 0; j < 8; ++j) s[j] = v8f_zero();
  const bf16* qrow = Qg + (base + w * 16) * EE + co;
#pragma unroll
  for (int ks = 0; ks < 64; ks += 32) {
    v16bf aq = load_frag(qrow + ks, EE);
#pragma unroll
    for (int j = 0; j < 8; ++j) {
      v16bf bk = load_frag(&Ks[j * 16 * 64 + ks], 64);
      s[j] = wmma_bf16(aq, bk, s[j]);
    }
  }

  // softmax (scale 1/sqrt(64)); vgpr i: lanes 0-15 -> row w*16+i,
  // lanes 16-31 -> row w*16+8+i.
  const float scale = 0.125f;
  float rsum[8];
#pragma unroll
  for (int i = 0; i < 8; ++i) {
    float mx = s[0][i];
#pragma unroll
    for (int j = 1; j < 8; ++j) mx = fmaxf(mx, s[j][i]);
#pragma unroll
    for (int o = 1; o < 16; o <<= 1) mx = fmaxf(mx, __shfl_xor(mx, o));
    float t = 0.f;
#pragma unroll
    for (int j = 0; j < 8; ++j) {
      float p = __expf((s[j][i] - mx) * scale);
      s[j][i] = p;
      t += p;
    }
#pragma unroll
    for (int o = 1; o < 16; o <<= 1) t += __shfl_xor(t, o);
    rsum[i] = t;
  }

  const int nl = lane & 15;
  const int mb = (lane >> 4) << 3;
#pragma unroll
  for (int j = 0; j < 8; ++j)
#pragma unroll
    for (int i = 0; i < 8; ++i)
      Ps[(w * 16 + mb + i) * 128 + j * 16 + nl] = (bf16)s[j][i];
  __syncthreads();

  // O = P @ V
  v8f o[4];
#pragma unroll
  for (int j = 0; j < 4; ++j) o[j] = v8f_zero();
#pragma unroll
  for (int ks = 0; ks < 128; ks += 32) {
    v16bf ap = load_frag(&Ps[(w * 16) * 128 + ks], 128);
#pragma unroll
    for (int j = 0; j < 4; ++j) {
      v16bf bv = load_frag(&Vt[j * 16 * 128 + ks], 128);
      o[j] = wmma_bf16(ap, bv, o[j]);
    }
  }
#pragma unroll
  for (int j = 0; j < 4; ++j)
#pragma unroll
    for (int i = 0; i < 8; ++i) {
      float ov = o[j][i] / rsum[i];
      Og[(base + w * 16 + mb + i) * EE + co + j * 16 + nl] = (bf16)ov;
    }
}

// ---------------------------------------------------------------------------
// final: out[m] = h[m,:2048] . w2[0,:2048] + b2  (one block per row)
// ---------------------------------------------------------------------------
__global__ __launch_bounds__(256) void final_kernel(const bf16* __restrict__ h,
                                                    const float* __restrict__ wv,
                                                    const float* __restrict__ bv,
                                                    float* __restrict__ out) {
  const int m   = blockIdx.x;
  const int tid = threadIdx.x;
  BF8 t;
  t.u4 = *(const uint4*)&h[(size_t)m * 2048 + tid * 8];
  const float* wr = wv + tid * 8;
  float s = 0.f;
#pragma unroll
  for (int i = 0; i < 8; ++i) s = fmaf((float)t.b[i], wr[i], s);
#pragma unroll
  for (int o = 1; o < 32; o <<= 1) s += __shfl_xor(s, o);
  __shared__ float red[8];
  if ((tid & 31) == 0) red[tid >> 5] = s;
  __syncthreads();
  if (tid == 0) {
    float ts = 0.f;
#pragma unroll
    for (int i = 0; i < 8; ++i) ts += red[i];
    out[m] = ts + bv[0];
  }
}

// ---------------------------------------------------------------------------
// host side
// ---------------------------------------------------------------------------
struct FFp  { const float *g0, *b0, *w1, *w2, *b2; };
struct MHAp { const float *win, *bin, *wout, *bout; };

extern "C" void kernel_launch(void* const* d_in, const int* in_sizes, int n_in,
                              void* d_out, int out_size, void* d_ws, size_t ws_size,
                              hipStream_t stream) {
  (void)in_sizes; (void)n_in; (void)out_size; (void)ws_size;
  const float* features = (const float*)d_in[0];
  // d_in[1] = ensemble_index (contiguous equal groups -> implied by layout)

  int pi = 2;
  auto nf = [&]() { return (const float*)d_in[pi++]; };
  auto ffp  = [&]() { FFp p;  p.g0 = nf(); p.b0 = nf(); p.w1 = nf(); p.w2 = nf(); p.b2 = nf(); return p; };
  auto mhap = [&]() { MHAp p; p.win = nf(); p.bin = nf(); p.wout = nf(); p.bout = nf(); return p; };
  FFp  l1kqv = ffp();  MHAp l1m = mhap();  FFp l1ff = ffp();
  FFp  l2kqv = ffp();  MHAp l2m = mhap();  FFp l2ff = ffp();
  FFp  fin   = ffp();

  size_t off = 0;
  auto walloc = [&](size_t bytes) -> void* {
    void* p = (char*)d_ws + off;
    off = (off + bytes + 255) & ~(size_t)255;
    return p;
  };
  auto cvt = [&](const float* src, size_t n) -> bf16* {
    bf16* dst = (bf16*)walloc(n * sizeof(bf16));
    int n4 = (int)(n / 4);
    cvt_kernel<<<(n4 + 255) / 256, 256, 0, stream>>>(src, dst, n4);
    return dst;
  };

  bf16 *l1w1b = cvt(l1kqv.w1, (size_t)2048 * 512);
  bf16 *l1w2b = cvt(l1kqv.w2, (size_t)1536 * 2048);
  bf16 *l1wib = cvt(l1m.win,  (size_t)1536 * 512);
  bf16 *l1wob = cvt(l1m.wout, (size_t)512 * 512);
  bf16 *l1f1b = cvt(l1ff.w1,  (size_t)2048 * 512);
  bf16 *l1f2b = cvt(l1ff.w2,  (size_t)512 * 2048);
  bf16 *l2w1b = cvt(l2kqv.w1, (size_t)2048 * 512);
  bf16 *l2w2b = cvt(l2kqv.w2, (size_t)1536 * 2048);
  bf16 *l2wib = cvt(l2m.win,  (size_t)1536 * 512);
  bf16 *l2wob = cvt(l2m.wout, (size_t)512 * 512);
  bf16 *l2f1b = cvt(l2ff.w1,  (size_t)2048 * 512);
  bf16 *l2f2b = cvt(l2ff.w2,  (size_t)512 * 2048);
  bf16 *fnw1b = cvt(fin.w1,   (size_t)2048 * 512);

  float* xA   = (float*)walloc((size_t)NN * EE * 4);
  float* xB   = (float*)walloc((size_t)NN * EE * 4);
  bf16*  lnb  = (bf16*) walloc((size_t)NN * EE * 2);
  bf16*  hb   = (bf16*) walloc((size_t)NN * 2048 * 2);
  bf16*  kqvb = (bf16*) walloc((size_t)NN * 1536 * 2);
  bf16*  qb   = (bf16*) walloc((size_t)NN * EE * 2);
  bf16*  kb   = (bf16*) walloc((size_t)NN * EE * 2);
  bf16*  vb   = (bf16*) walloc((size_t)NN * EE * 2);
  bf16*  ob   = (bf16*) walloc((size_t)NN * EE * 2);

  auto gemm = [&](int act, const bf16* A, int lda, const bf16* W, int ldw,
                  const float* bias, const float* resid, float rs,
                  float* oF, bf16* oB, int Nout, int K) {
    dim3 grid(Nout / 256, NN / 128);
    if (act)
      gemm_bf16_kernel<1><<<grid, 256, 0, stream>>>(A, lda, W, ldw, bias, resid, rs, oF, oB, Nout, K);
    else
      gemm_bf16_kernel<0><<<grid, 256, 0, stream>>>(A, lda, W, ldw, bias, resid, rs, oF, oB, Nout, K);
  };

  slice_kernel<<<(NN * 128) / 256, 256, 0, stream>>>(features, xA);

  auto layer = [&](const FFp& kqv, const MHAp& mha, const FFp& ff,
                   bf16* w1b, bf16* w2b, bf16* wib, bf16* wob,
                   bf16* f1b, bf16* f2b) {
    ln_kernel<<<NN, 256, 0, stream>>>(xA, kqv.g0, kqv.b0, lnb);
    gemm(1, lnb, EE, w1b, EE, nullptr, nullptr, 0.f, nullptr, hb, 2048, EE);
    gemm(0, hb, 2048, w2b, 2048, kqv.b2, nullptr, 0.f, nullptr, kqvb, 1536, 2048);
    // reference calls attention(k, q, v): query <- cols 0:512, key <- 512:1024,
    // value <- 1024:1536; weights wq/wk/wv are w_in row blocks in order.
    gemm(0, kqvb + 0,    1536, wib,                      EE, mha.bin,        nullptr, 0.f, nullptr, qb, EE, EE);
    gemm(0, kqvb + 512,  1536, wib + (size_t)512 * 512,  EE, mha.bin + 512,  nullptr, 0.f, nullptr, kb, EE, EE);
    gemm(0, kqvb + 1024, 1536, wib + (size_t)1024 * 512, EE, mha.bin + 1024, nullptr, 0.f, nullptr, vb, EE, EE);
    attn_kernel<<<(NN / GG) * HH, 256, 0, stream>>>(qb, kb, vb, ob);
    gemm(0, ob, EE, wob, EE, mha.bout, xA, 1.0f, xB, nullptr, EE, EE);   // + x
    ln_kernel<<<NN, 256, 0, stream>>>(xB, ff.g0, ff.b0, lnb);
    gemm(1, lnb, EE, f1b, EE, nullptr, nullptr, 0.f, nullptr, hb, 2048, EE);
    gemm(0, hb, 2048, f2b, 2048, ff.b2, xB, 2.0f, xA, nullptr, EE, 2048); // + 2*attn
  };

  layer(l1kqv, l1m, l1ff, l1w1b, l1w2b, l1wib, l1wob, l1f1b, l1f2b);
  layer(l2kqv, l2m, l2ff, l2w1b, l2w2b, l2wib, l2wob, l2f1b, l2f2b);

  ln_kernel<<<NN, 256, 0, stream>>>(xA, fin.g0, fin.b0, lnb);
  gemm(1, lnb, EE, fnw1b, EE, nullptr, nullptr, 0.f, nullptr, hb, 2048, EE);
  final_kernel<<<NN, 256, 0, stream>>>(hb, fin.w2, fin.b2, (float*)d_out);
}
